// AttentionNetwork_3032246911157
// MI455X (gfx1250) — matrix operand order repaired and checked by reading, not compiled, not run on hardware
//
#include <hip/hip_runtime.h>

// ---------------------------------------------------------------------------
// Problem constants (match reference)
// ---------------------------------------------------------------------------
#define NB 8
#define NSQ 1024
#define NSK 1024
#define ND 1024
#define NH 8
#define NDH 128

typedef __bf16 bf16_t;
typedef __attribute__((ext_vector_type(16))) __bf16 v16bf;
typedef __attribute__((ext_vector_type(8)))  float  v8f;

// ---------------------------------------------------------------------------
// CDNA5 async global->LDS path (probed; falls back to sync load+ds_store)
// ---------------------------------------------------------------------------
#if defined(__has_builtin)
#if __has_builtin(__builtin_amdgcn_global_load_async_to_lds_b128) && \
    __has_builtin(__builtin_amdgcn_s_wait_asynccnt)
#define USE_ASYNC_LDS 1
#endif
#endif
#ifndef USE_ASYNC_LDS
#define USE_ASYNC_LDS 0
#endif

#if USE_ASYNC_LDS
// Builtin expects: (v4i addrspace(1)* src, v4i addrspace(3)* dst, imm off, imm cpol)
typedef int v4i __attribute__((vector_size(4 * sizeof(int))));
typedef __attribute__((address_space(1))) v4i gv4i;
typedef __attribute__((address_space(3))) v4i lv4i;
#endif

__device__ __forceinline__ void wait_async_lds() {
#if USE_ASYNC_LDS
  __builtin_amdgcn_s_wait_asynccnt(0);
#endif
}

// ---------------------------------------------------------------------------
// Helpers
// ---------------------------------------------------------------------------
__device__ __forceinline__ v16bf pack_frag(uint4 c0, uint4 c1) {
  union Cvt { uint4 u; __bf16 h[8]; };
  Cvt a; a.u = c0;
  Cvt b; b.u = c1;
  v16bf r;
#pragma unroll
  for (int e = 0; e < 8; ++e) { r[e] = a.h[e]; r[e + 8] = b.h[e]; }
  return r;
}

constexpr int GLS = 48;  // padded LDS row stride (elems); keeps 16B alignment

// Load one 128x32 tile pair into LDS buffers (async on CDNA5).
__device__ __forceinline__ void tile_load(const bf16_t* __restrict__ gA,
                                          const bf16_t* __restrict__ gB,
                                          bf16_t* la, bf16_t* lb,
                                          int K, int k0, int lr0, int lc0) {
#pragma unroll
  for (int i = 0; i < 2; ++i) {
    const int r = lr0 + i * 64;
#if USE_ASYNC_LDS
    __builtin_amdgcn_global_load_async_to_lds_b128(
        (gv4i*)(gA + (size_t)r * K + k0 + lc0), (lv4i*)&la[r * GLS + lc0], 0, 0);
    __builtin_amdgcn_global_load_async_to_lds_b128(
        (gv4i*)(gB + (size_t)r * K + k0 + lc0), (lv4i*)&lb[r * GLS + lc0], 0, 0);
#else
    uint4 va = *(const uint4*)(gA + (size_t)r * K + k0 + lc0);
    uint4 vb = *(const uint4*)(gB + (size_t)r * K + k0 + lc0);
    *(uint4*)&la[r * GLS + lc0] = va;
    *(uint4*)&lb[r * GLS + lc0] = vb;
#endif
  }
}

// ---------------------------------------------------------------------------
// GEMM: C[M,N] = A[M,K](bf16,row) * Bt[N,K](bf16,row)^T  (+bias, relu, out cvt)
// 128x128 tile, K-step 32, 256 threads = 8 waves in 2x4 grid,
// double-buffered LDS with async global->LDS prefetch overlapping WMMA.
// ---------------------------------------------------------------------------
template <bool BIAS, bool RELU, bool OUT_BF16>
__global__ __launch_bounds__(256) void gemm_bf16_tn(
    const bf16_t* __restrict__ A, const bf16_t* __restrict__ Bt,
    const float* __restrict__ bias, void* __restrict__ Cout,
    int M, int N, int K) {
  constexpr int TM = 128, TN = 128, TK = 32;
  __shared__ __align__(16) bf16_t lA[2][TM * GLS];
  __shared__ __align__(16) bf16_t lB[2][TN * GLS];

  const int tiles_n = N / TN;
  const int tm = (int)blockIdx.x / tiles_n;
  const int tn = (int)blockIdx.x % tiles_n;
  const int tid  = threadIdx.x;
  const int lane = tid & 31;
  const int wv   = tid >> 5;
  const int wm   = wv >> 2;  // 0..1  (64 rows each)
  const int wn   = wv & 3;   // 0..3  (32 cols each)

  v8f acc[4][2];
#pragma unroll
  for (int i = 0; i < 4; ++i)
#pragma unroll
    for (int j = 0; j < 2; ++j)
#pragma unroll
      for (int e = 0; e < 8; ++e) acc[i][j][e] = 0.0f;

  const int lr0 = tid >> 2;       // rows 0..63 (+64 on second chunk)
  const int lc0 = (tid & 3) * 8;  // 8-elem (16B) column chunk

  const bf16_t* gA = A  + (size_t)(tm * TM) * K;
  const bf16_t* gB = Bt + (size_t)(tn * TN) * K;

  // fragment LDS addresses
  const int ar = wm * 64 + (lane & 15);
  const int ak = (lane >> 4) * 8;
  const int br = wn * 32 + (lane & 15);
  const int bk = (lane >> 4) * 8;

  tile_load(gA, gB, lA[0], lB[0], K, 0, lr0, lc0);
  int p = 0;
  for (int k0 = 0; k0 < K; k0 += TK) {
    wait_async_lds();       // own async stores for buf p landed
    __syncthreads();        // everyone's landed; everyone done reading buf p^1
    if (k0 + TK < K)        // prefetch next tile into the other buffer
      tile_load(gA, gB, lA[p ^ 1], lB[p ^ 1], K, k0 + TK, lr0, lc0);

    // A fragment: lane<16 -> K {0..7,16..23}, lane>=16 -> K {8..15,24..31}
    v16bf af[4], bfv[2];
#pragma unroll
    for (int i = 0; i < 4; ++i) {
      const bf16_t* q = &lA[p][(ar + i * 16) * GLS + ak];
      af[i] = pack_frag(*(const uint4*)q, *(const uint4*)(q + 16));
    }
    // B fragment: lane<16 -> K {0..7,16..23} of column n, lane>=16 -> K {8..15,24..31}
#pragma unroll
    for (int j = 0; j < 2; ++j) {
      const bf16_t* q = &lB[p][(br + j * 16) * GLS + bk];
      bfv[j] = pack_frag(*(const uint4*)q, *(const uint4*)(q + 16));
    }
#pragma unroll
    for (int i = 0; i < 4; ++i)
#pragma unroll
      for (int j = 0; j < 2; ++j)
        acc[i][j] = __builtin_amdgcn_wmma_f32_16x16x32_bf16(
            false, af[i], false, bfv[j], (short)0, acc[i][j], false, false);
    p ^= 1;
  }

  // Epilogue. C layout: VGPR r, lane<16 -> M=r ; lane>=16 -> M=r+8 ; N=lane&15.
  const int rh = (lane >> 4) * 8;
#pragma unroll
  for (int j = 0; j < 2; ++j) {
    const int col = tn * TN + wn * 32 + j * 16 + (lane & 15);
    float bv = 0.0f;
    if constexpr (BIAS) bv = bias[col];
#pragma unroll
    for (int i = 0; i < 4; ++i) {
      const int row0 = tm * TM + wm * 64 + i * 16 + rh;
#pragma unroll
      for (int r = 0; r < 8; ++r) {
        float v = acc[i][j][r];
        if constexpr (BIAS) v += bv;
        if constexpr (RELU) v = fmaxf(v, 0.0f);
        const size_t off = (size_t)(row0 + r) * N + col;
        if constexpr (OUT_BF16) ((bf16_t*)Cout)[off] = (bf16_t)v;
        else                    ((float*)Cout)[off]  = v;
      }
    }
  }
}

// ---------------------------------------------------------------------------
// W[K,N] f32 -> Wt[N,K] bf16 (transpose + convert), 32x32 tiles
// ---------------------------------------------------------------------------
__global__ __launch_bounds__(256) void transpose_to_bf16_kernel(
    const float* __restrict__ W, bf16_t* __restrict__ Wt, int K, int N) {
  __shared__ float tile[32][33];
  const int bx = (int)blockIdx.x % (N / 32);  // n tile
  const int by = (int)blockIdx.x / (N / 32);  // k tile
  const int x = threadIdx.x & 31, y = threadIdx.x >> 5;
#pragma unroll
  for (int yy = y; yy < 32; yy += 8)
    tile[yy][x] = W[(size_t)(by * 32 + yy) * N + bx * 32 + x];
  __syncthreads();
#pragma unroll
  for (int yy = y; yy < 32; yy += 8)
    Wt[(size_t)(bx * 32 + yy) * K + by * 32 + x] = (bf16_t)tile[x][yy];
}

// f32 -> bf16 elementwise
__global__ void f32_to_bf16_kernel(const float* __restrict__ in,
                                   bf16_t* __restrict__ out, int n) {
  const int i = blockIdx.x * blockDim.x + threadIdx.x;
  if (i < n) out[i] = (bf16_t)in[i];
}

// cos/sin of per-head scalar coordinates x[b,s]*W[h]; writes both
// [b,s,h] layout (for attout/Z lookup) and [b,h,s] layout (for Z / CvSv sweeps).
__global__ void trig_kernel(const float* __restrict__ xin,
                            const float* __restrict__ W,
                            float* __restrict__ cbsh, float* __restrict__ sbsh,
                            float* __restrict__ cbhs, float* __restrict__ sbhs,
                            int S) {
  const int i = blockIdx.x * blockDim.x + threadIdx.x;  // over B*S*H
  if (i >= NB * S * NH) return;
  const int h = i & (NH - 1);
  const int bs = i >> 3;
  const int b = bs / S;
  const int s = bs - b * S;
  const float v = xin[bs] * W[h];
  const float c = cosf(v), sn = sinf(v);
  cbsh[i] = c;
  sbsh[i] = sn;
  cbhs[(size_t)(b * NH + h) * S + s] = c;
  sbhs[(size_t)(b * NH + h) * S + s] = sn;
}

// Cv[b,h,d] = sum_k ck[b,h,k]*v[b,k,h*DH+d] ; Sv likewise. One block per (b,h).
__global__ __launch_bounds__(128) void cvsv_kernel(
    const float* __restrict__ ckT, const float* __restrict__ skT,
    const float* __restrict__ v, float* __restrict__ Cv, float* __restrict__ Sv) {
  const int bh = blockIdx.x;            // 0..63
  const int b = bh >> 3, h = bh & 7;
  const int d = threadIdx.x;            // 0..127
  const float* ck = ckT + (size_t)bh * NSK;
  const float* sk = skT + (size_t)bh * NSK;
  float ca = 0.f, sa = 0.f;
  for (int k = 0; k < NSK; ++k) {
    const float vv = v[(size_t)(b * NSK + k) * ND + h * NDH + d];
    ca = fmaf(ck[k], vv, ca);
    sa = fmaf(sk[k], vv, sa);
  }
  Cv[bh * NDH + d] = ca;
  Sv[bh * NDH + d] = sa;
}

// Z[b,q,h] = eps + sum_k |cq*ck + sq*sk|  (unit-stride float4 over [b,h,k])
__global__ __launch_bounds__(256) void z_kernel(
    const float* __restrict__ cq, const float* __restrict__ sq,
    const float* __restrict__ ckT, const float* __restrict__ skT,
    float* __restrict__ Z) {
  const int i = blockIdx.x * 256 + threadIdx.x;  // B*SQ*H, [b,q,h] layout
  if (i >= NB * NSQ * NH) return;
  const int h = i & (NH - 1);
  const int b = (i >> 3) >> 10;
  const float cqv = cq[i], sqv = sq[i];
  const float4* c4p = (const float4*)(ckT + (size_t)(b * NH + h) * NSK);
  const float4* s4p = (const float4*)(skT + (size_t)(b * NH + h) * NSK);
  float acc = 0.f;
  for (int k = 0; k < NSK / 4; ++k) {
    const float4 c4 = c4p[k], s4 = s4p[k];
    acc += fabsf(fmaf(cqv, c4.x, sqv * s4.x));
    acc += fabsf(fmaf(cqv, c4.y, sqv * s4.y));
    acc += fabsf(fmaf(cqv, c4.z, sqv * s4.z));
    acc += fabsf(fmaf(cqv, c4.w, sqv * s4.w));
  }
  Z[i] = acc + 1e-6f;
}

// out[b,q,h*DH+d] = (cq*Cv + sq*Sv)/Z  -> bf16 (feeds Wf GEMM)
__global__ void attout_kernel(const float* __restrict__ cq,
                              const float* __restrict__ sq,
                              const float* __restrict__ Cv,
                              const float* __restrict__ Sv,
                              const float* __restrict__ Z,
                              bf16_t* __restrict__ out) {
  const int i = blockIdx.x * 256 + threadIdx.x;  // B*SQ*D
  if (i >= NB * NSQ * ND) return;
  const int d  = i & (ND - 1);
  const int bq = i >> 10;
  const int h  = d >> 7;
  const int dd = d & (NDH - 1);
  const int b  = bq >> 10;
  const int qh = bq * NH + h;
  const int hd = (b * NH + h) * NDH + dd;
  const float v = (cq[qh] * Cv[hd] + sq[qh] * Sv[hd]) / Z[qh];
  out[i] = (bf16_t)v;
}

// LayerNorm over D=1024 of (a+b); optional bf16 shadow copy for next GEMM.
template <bool WRITE_BF16>
__global__ __launch_bounds__(256) void ln_kernel(
    const float* __restrict__ a, const float* __restrict__ b,
    const float* __restrict__ gamma, const float* __restrict__ beta,
    float* __restrict__ out_f32, bf16_t* __restrict__ out_bf) {
  __shared__ float red[16];
  const int row = blockIdx.x;
  const int lane = threadIdx.x & 31;
  const int wv = threadIdx.x >> 5;
  const size_t base = (size_t)row * ND;

  float vals[4];
  float s = 0.f;
#pragma unroll
  for (int i = 0; i < 4; ++i) {
    const int c = threadIdx.x + i * 256;
    vals[i] = a[base + c] + b[base + c];
    s += vals[i];
  }
#pragma unroll
  for (int o = 16; o > 0; o >>= 1) s += __shfl_xor(s, o, 32);
  if (lane == 0) red[wv] = s;
  __syncthreads();
  float tot = 0.f;
#pragma unroll
  for (int w = 0; w < 8; ++w) tot += red[w];
  const float mean = tot * (1.0f / ND);

  float vs = 0.f;
#pragma unroll
  for (int i = 0; i < 4; ++i) {
    const float d = vals[i] - mean;
    vs += d * d;
  }
#pragma unroll
  for (int o = 16; o > 0; o >>= 1) vs += __shfl_xor(vs, o, 32);
  if (lane == 0) red[8 + wv] = vs;
  __syncthreads();
  float tv = 0.f;
#pragma unroll
  for (int w = 0; w < 8; ++w) tv += red[8 + w];
  const float inv = rsqrtf(tv * (1.0f / ND) + 1e-5f);

#pragma unroll
  for (int i = 0; i < 4; ++i) {
    const int c = threadIdx.x + i * 256;
    const float o = (vals[i] - mean) * inv * gamma[c] + beta[c];
    out_f32[base + c] = o;
    if constexpr (WRITE_BF16) out_bf[base + c] = (bf16_t)o;
  }
}

// ---------------------------------------------------------------------------
// Host orchestration
// ---------------------------------------------------------------------------
extern "C" void kernel_launch(void* const* d_in, const int* in_sizes, int n_in,
                              void* d_out, int out_size, void* d_ws, size_t ws_size,
                              hipStream_t stream) {
  (void)in_sizes; (void)n_in; (void)out_size; (void)ws_size;
  const float* key_in   = (const float*)d_in[0];
  const float* query_in = (const float*)d_in[1];
  const float* value_in = (const float*)d_in[2];
  const float* residual = (const float*)d_in[3];
  const float* Wk  = (const float*)d_in[4];
  const float* Wq  = (const float*)d_in[5];
  const float* Wv  = (const float*)d_in[6];
  const float* Wf  = (const float*)d_in[7];
  const float* bfb = (const float*)d_in[8];
  const float* g1  = (const float*)d_in[9];
  const float* b1  = (const float*)d_in[10];
  const float* Wm1 = (const float*)d_in[11];
  const float* bm1 = (const float*)d_in[12];
  const float* Wm2 = (const float*)d_in[13];
  const float* bm2 = (const float*)d_in[14];
  const float* Wm3 = (const float*)d_in[15];
  const float* bm3 = (const float*)d_in[16];
  const float* g2  = (const float*)d_in[17];
  const float* b2  = (const float*)d_in[18];
  float* out = (float*)d_out;

  // Workspace arena
  char* ws = (char*)d_ws;
  size_t off = 0;
  auto alloc = [&](size_t bytes) -> void* {
    void* p = ws + off;
    off += (bytes + 255) & ~(size_t)255;
    return p;
  };
  const int M = NB * NSQ;  // 8192 rows for all big GEMMs
  float*  ck   = (float*)alloc(sizeof(float) * NB * NSK * NH);  // [b,s,h]
  float*  sk   = (float*)alloc(sizeof(float) * NB * NSK * NH);
  float*  cq   = (float*)alloc(sizeof(float) * NB * NSQ * NH);
  float*  sq   = (float*)alloc(sizeof(float) * NB * NSQ * NH);
  float*  ckT  = (float*)alloc(sizeof(float) * NB * NH * NSK);  // [b,h,s]
  float*  skT  = (float*)alloc(sizeof(float) * NB * NH * NSK);
  float*  cqT  = (float*)alloc(sizeof(float) * NB * NH * NSQ);
  float*  sqT  = (float*)alloc(sizeof(float) * NB * NH * NSQ);
  float*  Zb   = (float*)alloc(sizeof(float) * NB * NSQ * NH);
  float*  Cv   = (float*)alloc(sizeof(float) * NB * NH * NDH);
  float*  Sv   = (float*)alloc(sizeof(float) * NB * NH * NDH);
  bf16_t* val_bf = (bf16_t*)alloc(sizeof(bf16_t) * (size_t)M * ND);
  bf16_t* WvT  = (bf16_t*)alloc(sizeof(bf16_t) * (size_t)ND * ND);
  bf16_t* WfT  = (bf16_t*)alloc(sizeof(bf16_t) * (size_t)ND * ND);
  bf16_t* Wm1T = (bf16_t*)alloc(sizeof(bf16_t) * (size_t)ND * ND);
  bf16_t* Wm2T = (bf16_t*)alloc(sizeof(bf16_t) * (size_t)ND * ND);
  bf16_t* Wm3T = (bf16_t*)alloc(sizeof(bf16_t) * (size_t)ND * ND);
  float*  big0 = (float*)alloc(sizeof(float) * (size_t)M * ND);  // v -> y -> mlp_out
  bf16_t* att_bf = (bf16_t*)alloc(sizeof(bf16_t) * (size_t)M * ND);
  float*  x_f32  = (float*)alloc(sizeof(float) * (size_t)M * ND);
  bf16_t* x_bf   = (bf16_t*)alloc(sizeof(bf16_t) * (size_t)M * ND);
  bf16_t* h1_bf  = (bf16_t*)alloc(sizeof(bf16_t) * (size_t)M * ND);
  bf16_t* h2_bf  = (bf16_t*)alloc(sizeof(bf16_t) * (size_t)M * ND);

  const int elems = M * ND;  // 8388608
  const dim3 blk(256);
  const int gemm_grid = (M / 128) * (ND / 128);   // 512
  const int tr_grid   = (ND / 32) * (ND / 32);    // 1024

  // 1) trig coordinates (both layouts)
  trig_kernel<<<(NB * NSQ * NH) / 256, blk, 0, stream>>>(query_in, Wq, cq, sq, cqT, sqT, NSQ);
  trig_kernel<<<(NB * NSK * NH) / 256, blk, 0, stream>>>(key_in,   Wk, ck, sk, ckT, skT, NSK);

  // 2) conversions / transposes
  f32_to_bf16_kernel<<<elems / 256, blk, 0, stream>>>(value_in, val_bf, elems);
  transpose_to_bf16_kernel<<<tr_grid, blk, 0, stream>>>(Wv,  WvT,  ND, ND);
  transpose_to_bf16_kernel<<<tr_grid, blk, 0, stream>>>(Wf,  WfT,  ND, ND);
  transpose_to_bf16_kernel<<<tr_grid, blk, 0, stream>>>(Wm1, Wm1T, ND, ND);
  transpose_to_bf16_kernel<<<tr_grid, blk, 0, stream>>>(Wm2, Wm2T, ND, ND);
  transpose_to_bf16_kernel<<<tr_grid, blk, 0, stream>>>(Wm3, Wm3T, ND, ND);

  // 3) v = value_in @ Wv  (WMMA, f32 out into big0)
  gemm_bf16_tn<false, false, false><<<gemm_grid, blk, 0, stream>>>(
      val_bf, WvT, nullptr, big0, M, ND, ND);

  // 4) rank-2 attention pieces
  cvsv_kernel<<<NB * NH, dim3(128), 0, stream>>>(ckT, skT, big0, Cv, Sv);
  z_kernel<<<(NB * NSQ * NH) / 256, blk, 0, stream>>>(cq, sq, ckT, skT, Zb);
  attout_kernel<<<elems / 256, blk, 0, stream>>>(cq, sq, Cv, Sv, Zb, att_bf);

  // 5) y = att @ Wf + bf  (into big0, v no longer needed)
  gemm_bf16_tn<true, false, false><<<gemm_grid, blk, 0, stream>>>(
      att_bf, WfT, bfb, big0, M, ND, ND);

  // 6) x = LN(residual + y)  (f32 + bf16 shadow)
  ln_kernel<true><<<M, blk, 0, stream>>>(big0, residual, g1, b1, x_f32, x_bf);

  // 7) MLP
  gemm_bf16_tn<true, true, true><<<gemm_grid, blk, 0, stream>>>(
      x_bf, Wm1T, bm1, h1_bf, M, ND, ND);
  gemm_bf16_tn<true, true, true><<<gemm_grid, blk, 0, stream>>>(
      h1_bf, Wm2T, bm2, h2_bf, M, ND, ND);
  gemm_bf16_tn<true, false, false><<<gemm_grid, blk, 0, stream>>>(
      h2_bf, Wm3T, bm3, big0, M, ND, ND);  // mlp_out into big0

  // 8) final LN(x + mlp_out) -> d_out
  ln_kernel<false><<<M, blk, 0, stream>>>(x_f32, big0, g2, b2, out, nullptr);
}
